// GraphAndMessages_21964462751805
// MI455X (gfx1250) — matrix operand order repaired and checked by reading, not compiled, not run on hardware
//
#include <hip/hip_runtime.h>
#include <math.h>

typedef __attribute__((ext_vector_type(16))) _Float16 v16h;
typedef __attribute__((ext_vector_type(8)))  float    v8f;

#define BB 16
#define NN 256
#define DD 64

// Low 32 bits of a generic pointer to __shared__ == LDS byte offset (ISA 10.2).
__device__ __forceinline__ unsigned lds_off(const void* p) {
  return (unsigned)(uintptr_t)p;
}

// ---------------------------------------------------------------------------
// K1: per batch, P = X_b @ W1[:, :64]^T, Q = X_b @ W1[:, 64:]^T  (no bias: BN
// cancels b1 exactly). Also per-batch sums {SP, SQ, SP2, SQ2} for exact BN
// stats over all B*N*N edge rows.
// ---------------------------------------------------------------------------
__global__ __launch_bounds__(256) void k1_pq(const float* __restrict__ X,
                                             const float* __restrict__ W1,
                                             float* __restrict__ P,
                                             float* __restrict__ Qm,
                                             float* __restrict__ bstats) {
  __shared__ float sums[128];
  const int b = blockIdx.x, tid = threadIdx.x, lane = tid & 31;
  if (tid < 128) sums[tid] = 0.f;
  __syncthreads();

  const float4* xsrc = (const float4*)(X + ((size_t)b * NN + tid) * DD);
  float x[64];
#pragma unroll
  for (int i = 0; i < 16; i++) {
    float4 v = xsrc[i];
    x[4*i] = v.x; x[4*i+1] = v.y; x[4*i+2] = v.z; x[4*i+3] = v.w;
  }
  float* Prow = P  + ((size_t)b * NN + tid) * 32;
  float* Qrow = Qm + ((size_t)b * NN + tid) * 32;

  for (int j = 0; j < 32; j++) {
    float sp = 0.f, sq = 0.f;
#pragma unroll
    for (int k = 0; k < 64; k++) {           // W1 indices wave-uniform -> scalar loads
      sp = fmaf(x[k], W1[j * 128 + k],      sp);
      sq = fmaf(x[k], W1[j * 128 + 64 + k], sq);
    }
    Prow[j] = sp; Qrow[j] = sq;
    float vp = sp, vq = sq, vp2 = sp * sp, vq2 = sq * sq;
#pragma unroll
    for (int off = 16; off > 0; off >>= 1) {
      vp  += __shfl_xor(vp,  off); vq  += __shfl_xor(vq,  off);
      vp2 += __shfl_xor(vp2, off); vq2 += __shfl_xor(vq2, off);
    }
    if (lane == 0) {
      atomicAdd(&sums[j],      vp);  atomicAdd(&sums[32 + j], vq);
      atomicAdd(&sums[64 + j], vp2); atomicAdd(&sums[96 + j], vq2);
    }
  }
  __syncthreads();
  if (tid < 128) bstats[b * 128 + tid] = sums[tid];
}

// ---------------------------------------------------------------------------
// K2: fold BatchNorm1 into per-feature affine.
//   var = E[P^2]+E[Q^2]+2*E[PQ] - (E[P]+E[Q])^2,  E[PQ] = sum_b SP_b*SQ_b/(B*N*N)
//   scale = g1 * rsqrt(var+eps),  c = be1 - (E[P]+E[Q])*scale   (b1 cancels)
// ---------------------------------------------------------------------------
__global__ void k2_bn1(const float* __restrict__ bstats, const float* __restrict__ g1,
                       const float* __restrict__ be1, float* __restrict__ scale1,
                       float* __restrict__ c1) {
  int j = threadIdx.x;
  if (j >= 32) return;
  float SP = 0, SQ = 0, SP2 = 0, SQ2 = 0, CR = 0;
  for (int b = 0; b < BB; b++) {
    float sp = bstats[b * 128 + j], sq = bstats[b * 128 + 32 + j];
    SP += sp; SQ += sq;
    SP2 += bstats[b * 128 + 64 + j];
    SQ2 += bstats[b * 128 + 96 + j];
    CR += sp * sq;
  }
  const float invBN = 1.f / 4096.f;                 // 1/(B*N)
  float m0  = (SP + SQ) * invBN;
  float Eh2 = (SP2 + SQ2) * invBN + 2.f * CR * (1.f / (4096.f * 256.f));
  float var = Eh2 - m0 * m0;
  float sc  = g1[j] * rsqrtf(var + 1e-5f);
  scale1[j] = sc;
  c1[j]     = be1[j] - m0 * sc;
}

// K2b: P,Q are contiguous (2*131072 floats): apply per-feature scale in place.
__global__ __launch_bounds__(256) void k2b_scale(float* __restrict__ PQ,
                                                 const float* __restrict__ scale1) {
  int idx4 = blockIdx.x * 256 + threadIdx.x;        // 65536 float4 total
  float4 v  = ((float4*)PQ)[idx4];
  float4 sc = ((const float4*)scale1)[idx4 & 7];
  v.x *= sc.x; v.y *= sc.y; v.z *= sc.z; v.w *= sc.w;
  ((float4*)PQ)[idx4] = v;
}

// ---------------------------------------------------------------------------
// K3: 1M-edge MLP, transposed GEMM:  y^T = W2 @ h0^T.
//   A = W2 (f16 A-frags, built ONCE per wave; M = output feature)
//   B = h0^T (lane owns one edge column; 16 consecutive features per lane ->
//             4x ds_load_b128 from stride-36 LDS rows)
//   D: lane L holds 16 of 32 features of edge L%16; other 16 in lane L^16.
//   Layer-3 (32->2): 32 local FMAs + ONE shfl_xor(16) per output.
//   Aggregation: d = 16*(tile&3) + n0  -> 4 register accumulators, no LDS.
// Q'_b staged to LDS with CDNA5 async-to-LDS copies (pre-scaled by k2b).
// ---------------------------------------------------------------------------
__global__ __launch_bounds__(256) void k3_edges(
    const float* __restrict__ P, const float* __restrict__ Qm, const float* __restrict__ cvec,
    const float* __restrict__ W2, const float* __restrict__ b2,
    const float* __restrict__ W3, const float* __restrict__ b3,
    float* __restrict__ edges_out, float* __restrict__ agg) {
  __shared__ float Qs[NN * 36];                     // stride 36: 16B-aligned rows, bank spread
  const int b = blockIdx.x >> 5, sblk = blockIdx.x & 31;
  const int tid = threadIdx.x, wave = tid >> 5, lane = tid & 31;
  const int half = lane >> 4, n0 = lane & 15;
  const int s = sblk * 8 + wave;

  const float* Prow = P + ((size_t)b * NN + s) * 32 + half * 16;
  __builtin_prefetch(Prow, 0, 0);                   // global_prefetch_b8
  __builtin_prefetch(W2 + n0 * 32, 0, 0);

  {  // async-stage Q'_b (256x32 f32) into LDS: thread tid copies row tid
    const float* grow = Qm + ((size_t)b * NN + tid) * 32;
    unsigned dst = lds_off(&Qs[tid * 36]);
    asm volatile(
        "global_load_async_to_lds_b128 %0, %1, off\n\t"
        "global_load_async_to_lds_b128 %0, %1, off offset:16\n\t"
        "global_load_async_to_lds_b128 %0, %1, off offset:32\n\t"
        "global_load_async_to_lds_b128 %0, %1, off offset:48\n\t"
        "global_load_async_to_lds_b128 %0, %1, off offset:64\n\t"
        "global_load_async_to_lds_b128 %0, %1, off offset:80\n\t"
        "global_load_async_to_lds_b128 %0, %1, off offset:96\n\t"
        "global_load_async_to_lds_b128 %0, %1, off offset:112"
        :: "v"(dst), "v"(grow) : "memory");
    asm volatile("s_wait_asynccnt 0x0" ::: "memory");
  }

  // per-lane constants while the barrier settles -------------------------------
  // pc[i] = P'[s][16*half+i] + c[16*half+i]   (consecutive -> vector loads)
  float pc[16];
#pragma unroll
  for (int i = 0; i < 16; i++) pc[i] = Prow[i] + cvec[half * 16 + i];

  // A-frags for W2 (16x32 f16 layout): M = row = feature, K(i) = 16*half+i
  v16h a1, a2;
#pragma unroll
  for (int i = 0; i < 16; i++) {
    int K = (half << 4) + i;
    a1[i] = (_Float16)W2[n0 * 32 + K];              // features 0..15
    a2[i] = (_Float16)W2[(16 + n0) * 32 + K];       // features 16..31
  }
  // bias / W3 constants; this lane sees features f1 = r+8*half, f2 = 16+f1
  float b2a[8], b2b[8], w30A[8], w30B[8], w31A[8], w31B[8];
#pragma unroll
  for (int r = 0; r < 8; r++) {
    int f1 = r + 8 * half, f2 = 16 + f1;
    b2a[r] = b2[f1];        b2b[r] = b2[f2];
    w30A[r] = W3[f1];       w30B[r] = W3[f2];
    w31A[r] = W3[32 + f1];  w31B[r] = W3[32 + f2];
  }
  const float b30 = b3[0], b31 = b3[1];
  float* erow = edges_out + (size_t)b * 65536 + (size_t)s * 256;
  float agg4[4] = {0.f, 0.f, 0.f, 0.f};

  __syncthreads();                                  // LDS Q' visible to all waves

#pragma unroll
  for (int tile = 0; tile < 16; ++tile) {
    const int t0 = tile * 16;
    // B-frag: h0^T column for edge t0+n0; K = 16*half + i (consecutive)
    const float4* q4 = (const float4*)(&Qs[(t0 + n0) * 36 + half * 16]);
    float qv[16];
#pragma unroll
    for (int i = 0; i < 4; i++) {
      float4 v = q4[i];
      qv[4*i] = v.x; qv[4*i+1] = v.y; qv[4*i+2] = v.z; qv[4*i+3] = v.w;
    }
    v16h bf;
#pragma unroll
    for (int i = 0; i < 16; i++) {
      float x = pc[i] + qv[i];
      x = fmaxf(x, 0.01f * x);                      // leaky relu
      bf[i] = (_Float16)x;
    }
    v8f acc1, acc2;
#pragma unroll
    for (int r = 0; r < 8; r++) { acc1[r] = b2a[r]; acc2[r] = b2b[r]; }
    acc1 = __builtin_amdgcn_wmma_f32_16x16x32_f16(false, a1, false, bf, (short)0, acc1, false, false);
    acc2 = __builtin_amdgcn_wmma_f32_16x16x32_f16(false, a2, false, bf, (short)0, acc2, false, false);

    // layer-3: lane holds y[edge][f1_r] (acc1) and y[edge][f2_r] (acc2)
    float o0p = 0.f, o1p = 0.f;
#pragma unroll
    for (int r = 0; r < 8; r++) {
      float ya = acc1[r]; ya = fmaxf(ya, 0.01f * ya);
      float yb = acc2[r]; yb = fmaxf(yb, 0.01f * yb);
      o0p = fmaf(ya, w30A[r], o0p); o0p = fmaf(yb, w30B[r], o0p);
      o1p = fmaf(ya, w31A[r], o1p); o1p = fmaf(yb, w31B[r], o1p);
    }
    float o0 = o0p + __shfl_xor(o0p, 16);           // partner lane has other 16 feats
    float o1 = o1p + __shfl_xor(o1p, 16);
    float e  = 1.f / (1.f + __expf(-(o0 + b30)));   // sigmoid(edge logit)
    float w  = e * (o1 + b31);                      // weighted message
    if (half == 0) erow[t0 + n0] = e;               // 16-lane coalesced 64B store
    agg4[tile & 3] += w;                            // d = 16*(tile&3) + n0
  }

  if (half == 0) {
    float* ab = agg + ((size_t)b * NN + s) * DD;
    ab[n0]      = agg4[0];
    ab[16 + n0] = agg4[1];
    ab[32 + n0] = agg4[2];
    ab[48 + n0] = agg4[3];
  }
}

// ---------------------------------------------------------------------------
// K4: h2 = concat(X, agg) @ F1^T + fb1   (4096 x 16)
// ---------------------------------------------------------------------------
__global__ __launch_bounds__(256) void k4_final1(const float* __restrict__ X,
                                                 const float* __restrict__ agg,
                                                 const float* __restrict__ F1,
                                                 const float* __restrict__ fb1,
                                                 float* __restrict__ h2) {
  int r = blockIdx.x * 256 + threadIdx.x;
  float acc[16];
#pragma unroll
  for (int f = 0; f < 16; f++) acc[f] = fb1[f];
  const float4* xs = (const float4*)(X   + (size_t)r * DD);
  const float4* as = (const float4*)(agg + (size_t)r * DD);
  for (int kc = 0; kc < 8; kc++) {
    const float4* src = (kc < 4) ? (xs + kc * 4) : (as + (kc - 4) * 4);
    float c16[16];
#pragma unroll
    for (int i = 0; i < 4; i++) {
      float4 v = src[i];
      c16[4*i] = v.x; c16[4*i+1] = v.y; c16[4*i+2] = v.z; c16[4*i+3] = v.w;
    }
#pragma unroll
    for (int f = 0; f < 16; f++)
#pragma unroll
      for (int kk = 0; kk < 16; kk++)
        acc[f] = fmaf(c16[kk], F1[f * 128 + kc * 16 + kk], acc[f]);
  }
  float4* dst = (float4*)(h2 + (size_t)r * 16);
#pragma unroll
  for (int i = 0; i < 4; i++)
    dst[i] = make_float4(acc[4*i], acc[4*i+1], acc[4*i+2], acc[4*i+3]);
}

// K5: BN2 stats over 4096 rows x 16 feats -> folded affine (fb1 cancels).
__global__ __launch_bounds__(256) void k5_bn2(const float* __restrict__ h2,
                                              const float* __restrict__ fg,
                                              const float* __restrict__ fbb,
                                              float* __restrict__ scale2,
                                              float* __restrict__ c2) {
  __shared__ float acc[32];
  int tid = threadIdx.x, lane = tid & 31;
  if (tid < 32) acc[tid] = 0.f;
  __syncthreads();
  float a1[16] = {}, a2[16] = {};
  for (int i = 0; i < 16; i++) {
    const float* row = h2 + ((size_t)i * 256 + tid) * 16;
#pragma unroll
    for (int f = 0; f < 16; f++) { float v = row[f]; a1[f] += v; a2[f] += v * v; }
  }
#pragma unroll
  for (int f = 0; f < 16; f++) {
    float v1 = a1[f], v2 = a2[f];
#pragma unroll
    for (int off = 16; off > 0; off >>= 1) { v1 += __shfl_xor(v1, off); v2 += __shfl_xor(v2, off); }
    if (lane == 0) { atomicAdd(&acc[f], v1); atomicAdd(&acc[16 + f], v2); }
  }
  __syncthreads();
  if (tid < 16) {
    float mean = acc[tid] * (1.f / 4096.f);
    float var  = acc[16 + tid] * (1.f / 4096.f) - mean * mean;
    float sc   = fg[tid] * rsqrtf(var + 1e-5f);
    scale2[tid] = sc;
    c2[tid]     = fbb[tid] - mean * sc;
  }
}

// K6: pred = lrelu(bn(h2)) @ F2^T + fb2 -> (B,N,64)
__global__ __launch_bounds__(256) void k6_final2(const float* __restrict__ h2,
                                                 const float* __restrict__ scale2,
                                                 const float* __restrict__ c2,
                                                 const float* __restrict__ F2,
                                                 const float* __restrict__ fb2,
                                                 float* __restrict__ pred) {
  int tid = threadIdx.x;
  int row = blockIdx.x * 64 + (tid >> 2);
  int db  = (tid & 3) * 16;
  float a[16];
#pragma unroll
  for (int f = 0; f < 16; f++) {
    float v = h2[(size_t)row * 16 + f] * scale2[f] + c2[f];
    a[f] = fmaxf(v, 0.01f * v);
  }
  float outv[16];
#pragma unroll
  for (int i = 0; i < 16; i++) {
    int d = db + i;
    float s = fb2[d];
#pragma unroll
    for (int f = 0; f < 16; f++) s = fmaf(a[f], F2[d * 16 + f], s);
    outv[i] = s;
  }
  float4* dst = (float4*)(pred + (size_t)row * 64 + db);
#pragma unroll
  for (int i = 0; i < 4; i++)
    dst[i] = make_float4(outv[4*i], outv[4*i+1], outv[4*i+2], outv[4*i+3]);
}

// ---------------------------------------------------------------------------
extern "C" void kernel_launch(void* const* d_in, const int* in_sizes, int n_in,
                              void* d_out, int out_size, void* d_ws, size_t ws_size,
                              hipStream_t stream) {
  const float* X   = (const float*)d_in[0];
  const float* W1  = (const float*)d_in[1];
  const float* b1  = (const float*)d_in[2];   // cancels in BN (training mode)
  const float* g1  = (const float*)d_in[3];
  const float* be1 = (const float*)d_in[4];
  const float* W2  = (const float*)d_in[5];
  const float* b2  = (const float*)d_in[6];
  const float* W3  = (const float*)d_in[7];
  const float* b3  = (const float*)d_in[8];
  const float* F1  = (const float*)d_in[9];
  const float* fb1 = (const float*)d_in[10];
  const float* fg  = (const float*)d_in[11];
  const float* fbb = (const float*)d_in[12];
  const float* F2  = (const float*)d_in[13];
  const float* fb2 = (const float*)d_in[14];
  (void)b1; (void)in_sizes; (void)n_in; (void)out_size; (void)ws_size;

  float* out = (float*)d_out;
  float* ws  = (float*)d_ws;
  float* P      = ws;                 // 131072
  float* Q      = ws + 131072;        // 131072 (contiguous with P for k2b)
  float* bstats = ws + 262144;        // 2048
  float* scale1 = ws + 264192;        // 32
  float* c1     = ws + 264256;        // 32
  float* agg    = ws + 264320;        // 262144
  float* h2     = ws + 526464;        // 65536
  float* scale2 = ws + 592000;        // 16
  float* c2     = ws + 592064;        // 16
  float* edges  = out;                // B*E = 1048576
  float* pred   = out + 1048576;      // B*N*D = 262144

  k1_pq    <<<16,  256, 0, stream>>>(X, W1, P, Q, bstats);
  k2_bn1   <<<1,   32,  0, stream>>>(bstats, g1, be1, scale1, c1);
  k2b_scale<<<256, 256, 0, stream>>>(P, scale1);
  k3_edges <<<512, 256, 0, stream>>>(P, Q, c1, W2, b2, W3, b3, edges, agg);
  k4_final1<<<16,  256, 0, stream>>>(X, agg, F1, fb1, h2);
  k5_bn2   <<<1,   256, 0, stream>>>(h2, fg, fbb, scale2, c2);
  k6_final2<<<64,  256, 0, stream>>>(h2, scale2, c2, F2, fb2, pred);
}
